// ParameterizedHeuristicMatchingKnowledgeSelector_7816840479204
// MI455X (gfx1250) — matrix operand order repaired and checked by reading, not compiled, not run on hardware
//
#include <hip/hip_runtime.h>
#include <hip/hip_bf16.h>

typedef __attribute__((ext_vector_type(16))) __bf16 v16bf;
typedef __attribute__((ext_vector_type(8)))  __bf16 v8bf;
typedef __attribute__((ext_vector_type(4)))  __bf16 v4bf;
typedef __attribute__((ext_vector_type(8)))  float  v8f;
typedef __attribute__((ext_vector_type(4)))  float  f4;

#define Dn    1024
#define KTOK  512
#define BATCH 64
#define FDIM  4096
#define TOK   (BATCH * KTOK)   // 32768
#define XROW  514              // x is (64, 514, 1024)

#define MBLK  128              // tokens per workgroup (8 x 16 sub-tiles)
#define MT    (MBLK / 16)      // 8 sub-tiles per wave
#define NW    32               // output columns per wave (2 x 16)
#define NBLK  (8 * NW)         // 256 columns per workgroup
#define LROW  40               // padded LDS row stride in bf16 elems (80 B)

__device__ __forceinline__ float fast_tanh(float v) {
#if __has_builtin(__builtin_amdgcn_tanhf)
    return __builtin_amdgcn_tanhf(v);
#else
    return tanhf(v);
#endif
}

__device__ __forceinline__ v16bf cat16(v8bf lo, v8bf hi) {
    return __builtin_shufflevector(lo, hi,
        0,1,2,3,4,5,6,7,8,9,10,11,12,13,14,15);
}

__device__ __forceinline__ v8bf cat8(v4bf lo, v4bf hi) {
    return __builtin_shufflevector(lo, hi, 0,1,2,3,4,5,6,7);
}

// ---------------------------------------------------------------------------
// Kernel 1: zero partial-u accumulator in workspace
// ---------------------------------------------------------------------------
__global__ void zero_u_kernel(float* __restrict__ u) {
    int i = blockIdx.x * blockDim.x + threadIdx.x;
    if (i < TOK) u[i] = 0.0f;
}

// ---------------------------------------------------------------------------
// One feature section (SEC compile-time): 32 k-steps of 32 features each.
// SEC 0: z*q   1: z*m   2: |z-q|   3: |z-m|   (orow already selected)
// ---------------------------------------------------------------------------
template<int SEC>
__device__ __forceinline__ void gemm_section(
    const float* __restrict__ zrow,   // this thread's staging z row
    const float* __restrict__ orow,   // q-row or m-row (per SEC)
    const float* __restrict__ W,      // (4096,1024)
    __bf16 (*__restrict__ ldsA)[MBLK * LROW],
    int stok, int squad, int arow, int agrp, int lane, int n0w,
    v8f accA[MT], v8f accB[MT])
{
    #pragma unroll 2
    for (int cs = 0; cs < 32; ++cs) {
        __bf16* buf = ldsA[cs & 1];
        const int f0   = SEC * Dn + cs * 32;        // global feature base
        const int ccol = cs * 32 + squad * 16;      // column in D

        // ---- stage 128x32 bf16 feats tile (2 threads per token row) ----
        f4 z0 = *(const f4*)(zrow + ccol + 0);
        f4 z1 = *(const f4*)(zrow + ccol + 4);
        f4 z2 = *(const f4*)(zrow + ccol + 8);
        f4 z3 = *(const f4*)(zrow + ccol + 12);
        f4 o0 = *(const f4*)(orow + ccol + 0);
        f4 o1 = *(const f4*)(orow + ccol + 4);
        f4 o2 = *(const f4*)(orow + ccol + 8);
        f4 o3 = *(const f4*)(orow + ccol + 12);
        f4 r0, r1, r2, r3;
        if constexpr (SEC < 2) {
            r0 = z0 * o0; r1 = z1 * o1; r2 = z2 * o2; r3 = z3 * o3;
        } else {
            r0 = z0 - o0; r1 = z1 - o1; r2 = z2 - o2; r3 = z3 - o3;
            r0.x = fabsf(r0.x); r0.y = fabsf(r0.y); r0.z = fabsf(r0.z); r0.w = fabsf(r0.w);
            r1.x = fabsf(r1.x); r1.y = fabsf(r1.y); r1.z = fabsf(r1.z); r1.w = fabsf(r1.w);
            r2.x = fabsf(r2.x); r2.y = fabsf(r2.y); r2.z = fabsf(r2.z); r2.w = fabsf(r2.w);
            r3.x = fabsf(r3.x); r3.y = fabsf(r3.y); r3.z = fabsf(r3.z); r3.w = fabsf(r3.w);
        }
        __bf16* st = &buf[stok * LROW + squad * 16];
        *(v8bf*)(st + 0) = cat8(__builtin_convertvector(r0, v4bf),
                                __builtin_convertvector(r1, v4bf));
        *(v8bf*)(st + 8) = cat8(__builtin_convertvector(r2, v4bf),
                                __builtin_convertvector(r3, v4bf));
        __syncthreads();   // single barrier; ping-pong covers WAR hazard

        // ---- two B fragments: lane = N column (lane&15), 16 K in table order ----
        const float* wrow = W + (size_t)f0 * Dn + n0w + (lane & 15);
        __builtin_prefetch(wrow + 32 * Dn, 0, 0);   // next k-step's W row
        v16bf bfrag0, bfrag1;
        #pragma unroll
        for (int s = 0; s < 16; ++s) {
            const int kk = (s < 8) ? (agrp * 8 + s) : (16 + agrp * 8 + (s - 8));
            bfrag0[s] = (__bf16)wrow[(size_t)kk * Dn];
            bfrag1[s] = (__bf16)wrow[(size_t)kk * Dn + 16];
        }

        // ---- 8 M sub-tiles; each A fragment feeds 2 WMMAs (N halves) ----
        #pragma unroll
        for (int mt = 0; mt < MT; ++mt) {
            const __bf16* pa = &buf[(mt * 16 + arow) * LROW + agrp * 8];
            v8bf alo = *(const v8bf*)(pa);
            v8bf ahi = *(const v8bf*)(pa + 16);
            v16bf afrag = cat16(alo, ahi);
            accA[mt] = __builtin_amdgcn_wmma_f32_16x16x32_bf16(
                false, afrag, false, bfrag0, (short)0, accA[mt], false, false);
            accB[mt] = __builtin_amdgcn_wmma_f32_16x16x32_bf16(
                false, afrag, false, bfrag1, (short)0, accB[mt], false, false);
        }
    }
}

// ---------------------------------------------------------------------------
// Kernel 2: fused feats -> bf16 WMMA GEMM -> tanh -> dot(dot_bias) -> atomic u
// grid: (TOK/128, Dn/256), block: 256 threads (8 waves)
// wave w owns columns [blockIdx.y*256 + w*32, +32); block owns 128 tokens.
// ---------------------------------------------------------------------------
__global__ __launch_bounds__(256) void fused_gemm_kernel(
    const float* __restrict__ x,        // (64, 514, 1024)
    const float* __restrict__ W,        // (4096, 1024) row-major
    const float* __restrict__ dot_bias, // (1024,1) flat
    const float* __restrict__ bias1,    // (1024,)
    float* __restrict__ u)              // (32768,) partial accumulator
{
    __shared__ __align__(16) __bf16 ldsA[2][MBLK * LROW];   // 2 x 10240 B

    const int tid  = threadIdx.x;
    const int wave = tid >> 5;
    const int lane = tid & 31;
    const int t0   = blockIdx.x * MBLK;        // first token of M-block
    const int b    = t0 / KTOK;
    const int k0   = t0 % KTOK;                // 512 % 128 == 0 -> same batch
    const int n0w  = blockIdx.y * NBLK + wave * NW;

    // feats staging: 2 threads per token row, 16 floats each
    const int stok  = tid >> 1;                // 0..127
    const int squad = tid & 1;                 // 0..1
    const float* zrow = x + ((size_t)b * XROW + 2 + k0 + stok) * Dn;
    const float* qrow = x + ((size_t)b * XROW + 0) * Dn;
    const float* mrow = x + ((size_t)b * XROW + 1) * Dn;

    // A-fragment addressing (16-bit A 16x32 layout)
    const int arow = lane & 15;
    const int agrp = lane >> 4;

    v8f accA[MT] = {{}, {}, {}, {}, {}, {}, {}, {}};
    v8f accB[MT] = {{}, {}, {}, {}, {}, {}, {}, {}};

    gemm_section<0>(zrow, qrow, W, ldsA, stok, squad, arow, agrp, lane, n0w, accA, accB);
    gemm_section<1>(zrow, mrow, W, ldsA, stok, squad, arow, agrp, lane, n0w, accA, accB);
    gemm_section<2>(zrow, qrow, W, ldsA, stok, squad, arow, agrp, lane, n0w, accA, accB);
    gemm_section<3>(zrow, mrow, W, ldsA, stok, squad, arow, agrp, lane, n0w, accA, accB);

    // ---- epilogue: tanh(acc + bias1[n]) * dot_bias[n], fold both N halves,
    // then reduce the 16 lanes of each half-wave. C/D layout: vgpr i,
    // lanes 0-15 -> M=i, N=lane; lanes 16-31 -> M=8+i, N=lane-16.
    const int n    = n0w + (lane & 15);
    const int half = lane >> 4;
    const float b1a = bias1[n];
    const float dba = dot_bias[n];
    const float b1b = bias1[n + 16];
    const float dbb = dot_bias[n + 16];

    #pragma unroll
    for (int mt = 0; mt < MT; ++mt) {
        #pragma unroll
        for (int i = 0; i < 8; ++i) {
            float v = fast_tanh(accA[mt][i] + b1a) * dba
                    + fast_tanh(accB[mt][i] + b1b) * dbb;
            v += __shfl_xor(v, 1);
            v += __shfl_xor(v, 2);
            v += __shfl_xor(v, 4);
            v += __shfl_xor(v, 8);
            if ((lane & 15) == 0) {
                atomicAdd(&u[t0 + mt * 16 + half * 8 + i], v);
            }
        }
    }
}

// ---------------------------------------------------------------------------
// Kernel 3: softmax over k (512) per batch row. bias2 is uniform -> cancels.
// ---------------------------------------------------------------------------
__global__ __launch_bounds__(256) void softmax_kernel(
    const float* __restrict__ u, float* __restrict__ out)
{
    __shared__ float red[256];
    const int b   = blockIdx.x;
    const int tid = threadIdx.x;
    const float v0 = u[b * KTOK + tid];
    const float v1 = u[b * KTOK + 256 + tid];

    red[tid] = fmaxf(v0, v1);
    __syncthreads();
    for (int s = 128; s > 0; s >>= 1) {
        if (tid < s) red[tid] = fmaxf(red[tid], red[tid + s]);
        __syncthreads();
    }
    const float mx = red[0];
    __syncthreads();

    const float e0 = __expf(v0 - mx);
    const float e1 = __expf(v1 - mx);
    red[tid] = e0 + e1;
    __syncthreads();
    for (int s = 128; s > 0; s >>= 1) {
        if (tid < s) red[tid] += red[tid + s];
        __syncthreads();
    }
    const float inv = 1.0f / red[0];
    out[b * KTOK + tid]       = e0 * inv;
    out[b * KTOK + 256 + tid] = e1 * inv;
}

// ---------------------------------------------------------------------------
extern "C" void kernel_launch(void* const* d_in, const int* in_sizes, int n_in,
                              void* d_out, int out_size, void* d_ws, size_t ws_size,
                              hipStream_t stream) {
    const float* x        = (const float*)d_in[0];  // (64,514,1024)
    const float* W        = (const float*)d_in[1];  // (4096,1024)
    const float* dot_bias = (const float*)d_in[2];  // (1024,1)
    const float* bias1    = (const float*)d_in[3];  // (1024,)
    // d_in[4] = bias2: uniform additive constant -> exactly cancelled by softmax
    float* u   = (float*)d_ws;                      // 32768 floats of scratch
    float* out = (float*)d_out;                     // (64,512)

    zero_u_kernel<<<TOK / 256, 256, 0, stream>>>(u);

    dim3 grid(TOK / MBLK, Dn / NBLK);               // (256, 4)
    fused_gemm_kernel<<<grid, 256, 0, stream>>>(x, W, dot_bias, bias1, u);

    softmax_kernel<<<BATCH, 256, 0, stream>>>(u, out);
}
